// dann_5815385719236
// MI455X (gfx1250) — compile-verified, matrix operand order checked by reading, not run on hardware
//
#include <hip/hip_runtime.h>

typedef _Float16 v16h __attribute__((ext_vector_type(16)));
typedef _Float16 v8h  __attribute__((ext_vector_type(8)));
typedef float    v8f  __attribute__((ext_vector_type(8)));
typedef float    v4f  __attribute__((ext_vector_type(4)));

#define NB   256   // batch
#define NT   128   // time steps
#define NF   32    // features
#define NH   64    // hidden
#define G3   192   // 3*NH
#define HS   72    // padded LDS row stride for h (halves)

// Branch-free fast activations: exp->inf->rcp->0 gives correct saturation,
// so no EXEC-mask divergence near the WMMAs and ~10x fewer VALU ops than
// the IEEE divide / libm tanh expansions.
__device__ __forceinline__ float fast_sigmoid(float x) {
    return __builtin_amdgcn_rcpf(1.0f + __expf(-x));
}
__device__ __forceinline__ float fast_tanh(float x) {
    return 2.0f * __builtin_amdgcn_rcpf(1.0f + __expf(-2.0f * x)) - 1.0f;
}

__global__ void zero_common_kernel(float* __restrict__ common) {
    int i = blockIdx.x * blockDim.x + threadIdx.x;
    if (i < NB * NH) common[i] = 0.0f;
}

// One block per (feature, 32-batch tile). 8 waves of 32.
// Wave w: row tile r=w>>2 (16 batches), gate column base c=w&3 so it owns
// C-tiles (r,c), (r,c+4), (r,c+8) == gh_r, gh_z, gh_n for hidden cols 16c..16c+15.
__global__ __launch_bounds__(256)
void gru_kernel(const float* __restrict__ x,     // [B,T,F]
                const int*   __restrict__ lens,  // [B]
                const float* __restrict__ W_ih,  // [F,3H]
                const float* __restrict__ W_hh,  // [F,3H,H]
                const float* __restrict__ b_ih,  // [F,3H]
                const float* __restrict__ b_hh,  // [F,3H]
                float* __restrict__ common)      // [B,H] (atomic accum)
{
    __shared__ _Float16 hbuf[2][32 * HS];   // ping-pong f16 hidden state
    __shared__ float x_lds[NT * 32];        // [t][m] (transposed for vector reads)
    __shared__ float wih_s[G3], bih_s[G3], bhh_s[G3];
    __shared__ int   len_s[32];

    const int f      = blockIdx.x;
    const int batch0 = blockIdx.y * 32;
    const int tid  = threadIdx.x;
    const int lane = tid & 31;
    const int wv   = tid >> 5;
    const int la   = lane & 15;
    const int hh   = lane >> 4;
    const int r    = wv >> 2;   // 0..1
    const int c    = wv & 3;    // 0..3

    // ---- stage per-feature params, x tile (transposed), lens; zero h ----
    for (int i = tid; i < G3; i += 256) {
        wih_s[i] = W_ih[(size_t)f * G3 + i];
        bih_s[i] = b_ih[(size_t)f * G3 + i];
        bhh_s[i] = b_hh[(size_t)f * G3 + i];
    }
    if (tid < 32) len_s[tid] = lens[batch0 + tid];
    for (int i = tid; i < 32 * NT; i += 256) {
        int m = i >> 7, t = i & (NT - 1);
        x_lds[t * 32 + m] = x[((size_t)(batch0 + m) * NT + t) * NF + f];
    }
    for (int i = tid; i < 2 * 32 * HS; i += 256)
        ((_Float16*)hbuf)[i] = (_Float16)0.0f;

    // ---- B fragments (W_hh^T, f32->f16), resident in VGPRs all 128 steps ----
    // B[k,n] = W_hh[f, g0+n, k]; lane holds col n=la, K chunks {8h..8h+7, 16+8h..}
    v16h bfr[3][2];
    #pragma unroll
    for (int ti = 0; ti < 3; ++ti) {
        const int g = 16 * (c + 4 * ti) + la;
        const float* wp = W_hh + ((size_t)f * G3 + g) * NH;
        #pragma unroll
        for (int kc = 0; kc < 2; ++kc) {
            const float* p0 = wp + kc * 32 + 8 * hh;
            const float* p1 = p0 + 16;
            v16h bb;
            #pragma unroll
            for (int e = 0; e < 8; ++e) {
                bb[e]     = (_Float16)p0[e];
                bb[e + 8] = (_Float16)p1[e];
            }
            bfr[ti][kc] = bb;
        }
    }

    __syncthreads();

    // per-lane owned output elements (C layout): row 16r+8*hh+j, col 16c+la
    const int col   = 16 * c + la;
    const int mbase = 16 * r + 8 * hh;      // rows mbase..mbase+7 (consecutive)
    float hcur[8];
    int   len_j[8];
    #pragma unroll
    for (int j = 0; j < 8; ++j) { hcur[j] = 0.0f; len_j[j] = len_s[mbase + j]; }

    const float wr = wih_s[col],       br_ = bih_s[col],       cr_ = bhh_s[col];
    const float wz = wih_s[64 + col],  bz_ = bih_s[64 + col],  cz_ = bhh_s[64 + col];
    const float wn = wih_s[128 + col], bn_ = bih_s[128 + col], cn_ = bhh_s[128 + col];

    const int arow = 16 * r + la;           // A-fragment row this lane carries
    const float* xbase = &x_lds[mbase];     // + t*32 per step; 8 consecutive floats

    for (int t = 0; t < NT; ++t) {
        const int pp = t & 1;
        // ---- A fragments of h from LDS (two K=32 chunks covering H=64) ----
        const _Float16* hrow = &hbuf[pp][arow * HS];
        v16h a0, a1;
        {
            v8h l0v = *(const v8h*)(hrow + 8 * hh);
            v8h h0v = *(const v8h*)(hrow + 16 + 8 * hh);
            v8h l1v = *(const v8h*)(hrow + 32 + 8 * hh);
            v8h h1v = *(const v8h*)(hrow + 48 + 8 * hh);
            #pragma unroll
            for (int e = 0; e < 8; ++e) {
                a0[e] = l0v[e]; a0[e + 8] = h0v[e];
                a1[e] = l1v[e]; a1[e + 8] = h1v[e];
            }
        }
        // ---- x values for this lane's 8 rows: two b128 broadcast loads ----
        v4f xa = *(const v4f*)(xbase + t * 32);
        v4f xb = *(const v4f*)(xbase + t * 32 + 4);

        // ---- gh = h @ W_hh^T tiles: r, z, n (each K=64 -> 2 chained WMMAs) ----
        v8f accR = {0.f,0.f,0.f,0.f,0.f,0.f,0.f,0.f};
        v8f accZ = accR, accN = accR;
        accR = __builtin_amdgcn_wmma_f32_16x16x32_f16(false, a0, false, bfr[0][0], (short)0, accR, false, false);
        accR = __builtin_amdgcn_wmma_f32_16x16x32_f16(false, a1, false, bfr[0][1], (short)0, accR, false, false);
        accZ = __builtin_amdgcn_wmma_f32_16x16x32_f16(false, a0, false, bfr[1][0], (short)0, accZ, false, false);
        accZ = __builtin_amdgcn_wmma_f32_16x16x32_f16(false, a1, false, bfr[1][1], (short)0, accZ, false, false);
        accN = __builtin_amdgcn_wmma_f32_16x16x32_f16(false, a0, false, bfr[2][0], (short)0, accN, false, false);
        accN = __builtin_amdgcn_wmma_f32_16x16x32_f16(false, a1, false, bfr[2][1], (short)0, accN, false, false);

        // ---- gates + masked state update, mirror to next h buffer ----
        _Float16* hnew = &hbuf[pp ^ 1][mbase * HS + col];
        #pragma unroll
        for (int j = 0; j < 8; ++j) {
            const float xv = (j < 4) ? xa[j] : xb[j - 4];
            const float gr = xv * wr + br_ + accR[j] + cr_;
            const float gz = xv * wz + bz_ + accZ[j] + cz_;
            const float rg = fast_sigmoid(gr);
            const float zg = fast_sigmoid(gz);
            const float ng = fast_tanh(xv * wn + bn_ + rg * (accN[j] + cn_));
            const float hn = (1.0f - zg) * ng + zg * hcur[j];
            hcur[j] = (t < len_j[j]) ? hn : hcur[j];
            hnew[j * HS] = (_Float16)hcur[j];
        }
        __syncthreads();
    }

    // ---- feature-sum into common via f32 atomics ----
    #pragma unroll
    for (int j = 0; j < 8; ++j)
        atomicAdd(&common[(size_t)(batch0 + mbase + j) * NH + col], hcur[j]);
}

// Domain-classifier head: Linear -> BN(train stats) -> ReLU -> Linear -> log_softmax.
// Single block (needs batch-wide stats); 256 threads = one batch row each.
__global__ __launch_bounds__(256)
void head_kernel(const float* __restrict__ common,
                 const float* __restrict__ dc_w1, const float* __restrict__ dc_b1,
                 const float* __restrict__ gamma, const float* __restrict__ beta,
                 const float* __restrict__ dc_w2, const float* __restrict__ dc_b2,
                 float* __restrict__ out)
{
    __shared__ float w1[NH * NH];
    __shared__ float b1s[NH], gm[NH], bt[NH];
    __shared__ float w2s[2 * NH];
    __shared__ float ssum[NH], ssum2[NH];
    __shared__ float mu[NH], iv[NH];

    const int tid = threadIdx.x;
    for (int i = tid; i < NH * NH; i += 256) w1[i] = dc_w1[i];
    if (tid < NH) {
        b1s[tid] = dc_b1[tid];
        gm[tid]  = gamma[tid];
        bt[tid]  = beta[tid];
        ssum[tid] = 0.0f; ssum2[tid] = 0.0f;
    }
    if (tid < 2 * NH) w2s[tid] = dc_w2[tid];
    __syncthreads();

    float row[NH];
    #pragma unroll
    for (int k = 0; k < NH; ++k) row[k] = common[(size_t)tid * NH + k];

    float y[NH];
    for (int j = 0; j < NH; ++j) {
        float s = b1s[j];
        const float* wj = &w1[j * NH];
        #pragma unroll
        for (int k = 0; k < NH; ++k) s += row[k] * wj[k];
        y[j] = s;
        atomicAdd(&ssum[j],  s);
        atomicAdd(&ssum2[j], s * s);
    }
    __syncthreads();
    if (tid < NH) {
        const float m = ssum[tid] * (1.0f / NB);
        const float v = ssum2[tid] * (1.0f / NB) - m * m;   // biased var
        mu[tid] = m;
        iv[tid] = rsqrtf(v + 1e-5f);
    }
    __syncthreads();

    float l0 = dc_b2[0], l1 = dc_b2[1];
    for (int j = 0; j < NH; ++j) {
        float yn = (y[j] - mu[j]) * iv[j] * gm[j] + bt[j];
        yn = fmaxf(yn, 0.0f);
        l0 += yn * w2s[j];
        l1 += yn * w2s[NH + j];
    }
    const float mx  = fmaxf(l0, l1);
    const float lse = mx + __logf(__expf(l0 - mx) + __expf(l1 - mx));
    out[tid * 2 + 0] = l0 - lse;
    out[tid * 2 + 1] = l1 - lse;
}

extern "C" void kernel_launch(void* const* d_in, const int* in_sizes, int n_in,
                              void* d_out, int out_size, void* d_ws, size_t ws_size,
                              hipStream_t stream)
{
    const float* input = (const float*)d_in[0];
    /* d_in[1] input_diff: unused on this path */
    const int*   lens  = (const int*)  d_in[2];
    /* d_in[3] alpha: grad_reverse is identity in forward */
    const float* W_ih  = (const float*)d_in[4];
    const float* W_hh  = (const float*)d_in[5];
    const float* b_ih  = (const float*)d_in[6];
    const float* b_hh  = (const float*)d_in[7];
    const float* dc_w1 = (const float*)d_in[8];
    const float* dc_b1 = (const float*)d_in[9];
    const float* gamma = (const float*)d_in[10];
    const float* beta  = (const float*)d_in[11];
    const float* dc_w2 = (const float*)d_in[12];
    const float* dc_b2 = (const float*)d_in[13];

    float* out    = (float*)d_out;
    float* common = (float*)d_ws;              // [256,64] f32 scratch (64 KB)

    zero_common_kernel<<<dim3((NB * NH + 255) / 256), dim3(256), 0, stream>>>(common);
    gru_kernel<<<dim3(NF, NB / 32), dim3(256), 0, stream>>>(
        input, lens, W_ih, W_hh, b_ih, b_hh, common);
    head_kernel<<<dim3(1), dim3(256), 0, stream>>>(
        common, dc_w1, dc_b1, gamma, beta, dc_w2, dc_b2, out);
}
